// TransparentHybrid_75892072120385
// MI455X (gfx1250) — matrix operand order repaired and checked by reading, not compiled, not run on hardware
//
#include <hip/hip_runtime.h>

// ---------------------------------------------------------------------------
// Types for CDNA5 WMMA (gfx1250, wave32)
// ---------------------------------------------------------------------------
typedef __bf16 bf16;
typedef float  v8f   __attribute__((ext_vector_type(8)));
typedef __bf16 v16bf __attribute__((ext_vector_type(16)));
typedef __bf16 v8bf  __attribute__((ext_vector_type(8)));

union BFrag { v16bf v; v8bf h[2]; };

__device__ __forceinline__ bf16 f2bf(float f) {
    unsigned u = __float_as_uint(f);
    u += 0x7FFFu + ((u >> 16) & 1u);           // round-to-nearest-even
    unsigned short s = (unsigned short)(u >> 16);
    bf16 r; __builtin_memcpy(&r, &s, sizeof(r));
    return r;
}

__device__ __forceinline__ v8f wmma_bf16(const BFrag& a, const BFrag& b, v8f c) {
    // D = A(16x32 bf16) * B(32x16 bf16) + C(16x16 f32)
    return __builtin_amdgcn_wmma_f32_16x16x32_bf16(false, a.v, false, b.v,
                                                   (short)0, c, false, false);
}

// ---------------------------------------------------------------------------
// Generic WMMA GEMM: C[M,N] = A[M,K](bf16,row-major) * Bt[N,K](bf16,row-major)^T
// Block tile 256(M) x 64(N), K step 32, 8 waves, wave tile 64x32 (8 WMMAs/step).
// Ping-pong LDS buffers -> ONE barrier per K step; global prefetch of step s+1
// overlaps the 8 WMMAs of step s.
// Optional: fp32 out, bf16 out, fp32 residual add, ReLU, batched (b,h) offsets.
// Requires M%256==0, N%64==0, K%32==0 (true for every call in this model).
// ---------------------------------------------------------------------------
__global__ __launch_bounds__(256) void gemm_bf16_wmma(
    const bf16* __restrict__ A, const bf16* __restrict__ Bt,
    float* __restrict__ C, bf16* __restrict__ Cbf, const float* __restrict__ Radd,
    int M, int N, int K, int lda, int ldb, int ldc,
    long long sAb, long long sAh, long long sBb, long long sBh,
    long long sCb, long long sCh, int subH, int relu)
{
    __shared__ __align__(16) bf16 As[2][256][40];
    __shared__ __align__(16) bf16 Bs[2][64][40];

    const int tid  = threadIdx.x;
    const int lane = tid & 31, w = tid >> 5;
    const int wm = w & 3, wn = w >> 2;
    const int half = lane >> 4, l16 = lane & 15;
    const int m0 = blockIdx.y * 256, n0 = blockIdx.x * 64;

    long long offA = 0, offB = 0, offC = 0;
    if (subH > 0) {
        int z = blockIdx.z; int bb = z / subH, hh = z % subH;
        offA = bb * sAb + hh * sAh;
        offB = bb * sBb + hh * sBh;
        offC = bb * sCb + hh * sCh;
    }
    const bf16* Ap = A + offA;
    const bf16* Bp = Bt + offB;

    // per-thread staging slots for the cooperative tile copy
    const int rA = tid >> 2, cA = (tid & 3) * 8;   // + p*64 rows per pass
    const int rB = tid >> 2, cB = (tid & 3) * 8;

    v8bf stA[4], stB;
    auto loadG = [&](int k0) {
        #pragma unroll
        for (int p = 0; p < 4; ++p)
            stA[p] = *(const v8bf*)(Ap + (long long)(m0 + rA + p * 64) * lda + k0 + cA);
        stB = *(const v8bf*)(Bp + (long long)(n0 + rB) * ldb + k0 + cB);
    };
    auto storeL = [&](int buf) {
        #pragma unroll
        for (int p = 0; p < 4; ++p)
            *(v8bf*)&As[buf][rA + p * 64][cA] = stA[p];
        *(v8bf*)&Bs[buf][rB][cB] = stB;
    };

    v8f acc[4][2];
    #pragma unroll
    for (int i = 0; i < 4; ++i)
        #pragma unroll
        for (int j = 0; j < 2; ++j)
            for (int e = 0; e < 8; ++e) acc[i][j][e] = 0.f;

    const int nsteps = K / 32;
    loadG(0);
    storeL(0);
    __syncthreads();

    for (int s = 0; s < nsteps; ++s) {
        const int cur = s & 1, nxt = cur ^ 1;
        const bool more = (s + 1 < nsteps);
        if (more) loadG((s + 1) * 32);

        BFrag a[4], b[2];
        #pragma unroll
        for (int mi = 0; mi < 4; ++mi) {           // A frag: lane=M row, K halves
            int row = wm * 64 + mi * 16 + l16;
            a[mi].h[0] = *(const v8bf*)&As[cur][row][8 * half];
            a[mi].h[1] = *(const v8bf*)&As[cur][row][16 + 8 * half];
        }
        #pragma unroll
        for (int ni = 0; ni < 2; ++ni) {           // B frag: lane=N row, K halves
            int row = wn * 32 + ni * 16 + l16;
            b[ni].h[0] = *(const v8bf*)&Bs[cur][row][8 * half];
            b[ni].h[1] = *(const v8bf*)&Bs[cur][row][16 + 8 * half];
        }
        #pragma unroll
        for (int mi = 0; mi < 4; ++mi)
            #pragma unroll
            for (int ni = 0; ni < 2; ++ni)
                acc[mi][ni] = wmma_bf16(a[mi], b[ni], acc[mi][ni]);

        if (more) storeL(nxt);
        __syncthreads();
    }

    // epilogue: C layout VGPR r -> M = r + 8*half, N = l16
    #pragma unroll
    for (int mi = 0; mi < 4; ++mi)
        #pragma unroll
        for (int ni = 0; ni < 2; ++ni) {
            int rm = m0 + wm * 64 + mi * 16;
            int cn = n0 + wn * 32 + ni * 16 + l16;
            #pragma unroll
            for (int r = 0; r < 8; ++r) {
                int row = rm + r + 8 * half;
                long long off = offC + (long long)row * ldc + cn;
                float vv = acc[mi][ni][r];
                if (Radd) vv += Radd[off];
                if (relu) vv = fmaxf(vv, 0.f);
                if (C)    C[off]   = vv;
                if (Cbf)  Cbf[off] = f2bf(vv);
            }
        }
}

// ---------------------------------------------------------------------------
// Sliding-window flash attention with WMMA (window 512, head dim 64)
// One wave per 16 query rows. QK^T and P*V on v_wmma; P re-layout via LDS
// bounce guarded with s_wait_dscnt (intra-wave, in-order LDS).
// Writes bf16 directly (input to the o-projection GEMM).
// ---------------------------------------------------------------------------
__global__ __launch_bounds__(128) void swa_attn(
    const bf16* __restrict__ qb, const bf16* __restrict__ kb,
    const bf16* __restrict__ vt, bf16* __restrict__ outp,
    int T, int H, float scale)
{
    __shared__ __align__(16) bf16 pbuf[4][16][32];
    const int lane = threadIdx.x & 31, w = threadIdx.x >> 5;
    const int half = lane >> 4, l16 = lane & 15;
    const int b = blockIdx.z, h = blockIdx.y;
    const int q0b = blockIdx.x * 64;
    const int q0 = q0b + w * 16;
    const int HD = H * 64;

    BFrag aq[2];
    {
        const bf16* qrow = qb + (long long)(b * T + q0 + l16) * HD + h * 64;
        #pragma unroll
        for (int ks = 0; ks < 2; ++ks) {
            aq[ks].h[0] = *(const v8bf*)(qrow + ks * 32 + 8 * half);
            aq[ks].h[1] = *(const v8bf*)(qrow + ks * 32 + 16 + 8 * half);
        }
    }

    float m[8], l[8];
    v8f acc[4];
    #pragma unroll
    for (int r = 0; r < 8; ++r) { m[r] = -1e30f; l[r] = 0.f; }
    #pragma unroll
    for (int nf = 0; nf < 4; ++nf)
        for (int e = 0; e < 8; ++e) acc[nf][e] = 0.f;

    const int js = (q0b >= 512) ? (q0b - 512) : 0;   // uniform per block
    const int je = q0b + 63;
    for (int jc = js; jc <= je; jc += 32) {
        // ---- K fragments (Bt layout: lane = key, d runs at 8*half and 16+8*half) ----
        BFrag bk[2][2];
        #pragma unroll
        for (int kf = 0; kf < 2; ++kf) {
            int keyt = jc + 16 * kf + l16; if (keyt > T - 1) keyt = T - 1;
            const bf16* krow = kb + (long long)(b * T + keyt) * HD + h * 64;
            #pragma unroll
            for (int ks = 0; ks < 2; ++ks) {
                bk[kf][ks].h[0] = *(const v8bf*)(krow + ks * 32 + 8 * half);
                bk[kf][ks].h[1] = *(const v8bf*)(krow + ks * 32 + 16 + 8 * half);
            }
        }
        // ---- scores = Q * K^T (hd = 64 -> two k-steps) ----
        v8f sc[2];
        #pragma unroll
        for (int kf = 0; kf < 2; ++kf) {
            for (int e = 0; e < 8; ++e) sc[kf][e] = 0.f;
            sc[kf] = wmma_bf16(aq[0], bk[kf][0], sc[kf]);
            sc[kf] = wmma_bf16(aq[1], bk[kf][1], sc[kf]);
        }
        // ---- causal + window mask, streaming softmax ----
        float ps[2][8], pv[2][8];
        #pragma unroll
        for (int kf = 0; kf < 2; ++kf) {
            int key = jc + 16 * kf + l16;
            #pragma unroll
            for (int r = 0; r < 8; ++r) {
                int qr = q0 + r + 8 * half;
                float s = sc[kf][r] * scale;
                bool bad = (key > qr) || (key + 512 < qr);
                ps[kf][r] = bad ? -1e30f : s;
            }
        }
        #pragma unroll
        for (int r = 0; r < 8; ++r) {
            float mc = fmaxf(ps[0][r], ps[1][r]);
            #pragma unroll
            for (int msk = 1; msk < 16; msk <<= 1)
                mc = fmaxf(mc, __shfl_xor(mc, msk, 32));
            float mn = fmaxf(m[r], mc);
            float sf = __expf(m[r] - mn);
            float p0 = __expf(ps[0][r] - mn);
            float p1 = __expf(ps[1][r] - mn);
            float rs = p0 + p1;
            #pragma unroll
            for (int msk = 1; msk < 16; msk <<= 1)
                rs += __shfl_xor(rs, msk, 32);
            l[r] = l[r] * sf + rs;
            m[r] = mn;
            #pragma unroll
            for (int nf = 0; nf < 4; ++nf) acc[nf][r] *= sf;
            pv[0][r] = p0; pv[1][r] = p1;
        }
        // ---- P -> bf16 via per-wave LDS bounce (C layout -> A layout) ----
        #pragma unroll
        for (int r = 0; r < 8; ++r) {
            pbuf[w][r + 8 * half][l16]      = f2bf(pv[0][r]);
            pbuf[w][r + 8 * half][16 + l16] = f2bf(pv[1][r]);
        }
        asm volatile("s_wait_dscnt 0x0" ::: "memory");
        BFrag ap;
        ap.h[0] = *(const v8bf*)&pbuf[w][l16][8 * half];
        ap.h[1] = *(const v8bf*)&pbuf[w][l16][16 + 8 * half];
        // ---- out += P(16x32) * V(32x64): V^T rows are [d][t] ----
        #pragma unroll
        for (int nf = 0; nf < 4; ++nf) {
            const bf16* vrow = vt +
                (long long)((b * H + h) * 64 + nf * 16 + l16) * T + jc + 8 * half;
            BFrag bv;
            bv.h[0] = *(const v8bf*)(vrow);
            bv.h[1] = *(const v8bf*)(vrow + 16);
            acc[nf] = wmma_bf16(ap, bv, acc[nf]);
        }
    }
    // ---- normalize + store (bf16) ----
    #pragma unroll
    for (int nf = 0; nf < 4; ++nf)
        #pragma unroll
        for (int r = 0; r < 8; ++r) {
            int qr = q0 + r + 8 * half;
            outp[(long long)(b * T + qr) * HD + h * 64 + nf * 16 + l16] =
                f2bf(acc[nf][r] / l[r]);
        }
}

// ---------------------------------------------------------------------------
// Gated delta-rule scan. One block per (b,h), 64 threads (thread = V column),
// S[64] (K index) in registers. Writes recurrent outputs (bf16) and final
// state transposed S^T[v][k] in bf16 (= Bt layout for the retrieval GEMM).
// ---------------------------------------------------------------------------
__global__ __launch_bounds__(64) void gdn_scan(
    const float* __restrict__ kn, const float* __restrict__ vr,
    const float* __restrict__ beta, const float* __restrict__ gv,
    bf16* __restrict__ outs, bf16* __restrict__ Stb, int T)
{
    const int bh = blockIdx.x, b = bh >> 4, h = bh & 15;
    const int v = threadIdx.x;
    __shared__ float kt[64];
    float S[64];
    #pragma unroll
    for (int k = 0; k < 64; ++k) S[k] = 0.f;

    for (int t = 0; t < T; ++t) {
        long long row = (long long)b * T + t;
        long long base = row * 1024 + h * 64;
        kt[v] = kn[base + v];
        float vt = vr[base + v];
        float bt = beta[row * 16 + h];
        float gt = gv[row * 16 + h];
        __syncthreads();
        float pred = 0.f;
        #pragma unroll
        for (int k = 0; k < 64; ++k) pred += S[k] * kt[k];
        float com = bt * (vt - pred);
        float out = 0.f;
        #pragma unroll
        for (int k = 0; k < 64; ++k) {
            float sn = gt * S[k] + com * kt[k];
            S[k] = sn;
            out += sn * kt[k];
        }
        outs[base + v] = f2bf(out);
        __syncthreads();
    }
    long long sb = ((long long)bh * 64 + v) * 64;
    #pragma unroll
    for (int k = 0; k < 64; ++k) Stb[sb + k] = f2bf(S[k]);
}

// ---------------------------------------------------------------------------
// Small helper kernels
// ---------------------------------------------------------------------------
__global__ __launch_bounds__(256) void transpose_cast(
    const float* __restrict__ W, bf16* __restrict__ Wt, int K, int N)
{
    __shared__ float tile[32][33];
    int kb = blockIdx.x * 32, nb = blockIdx.y * 32;
    for (int i = threadIdx.y; i < 32; i += 8) {
        int k = kb + i, n = nb + threadIdx.x;
        tile[i][threadIdx.x] = (k < K && n < N) ? W[(long long)k * N + n] : 0.f;
    }
    __syncthreads();
    for (int i = threadIdx.y; i < 32; i += 8) {
        int n = nb + i, k = kb + threadIdx.x;
        if (n < N && k < K) Wt[(long long)n * K + k] = f2bf(tile[threadIdx.x][i]);
    }
}

__global__ void cast_f32_bf16(const float* __restrict__ in, bf16* __restrict__ out, long long n)
{
    long long i = (long long)blockIdx.x * blockDim.x + threadIdx.x;
    if (i < n) out[i] = f2bf(in[i]);
}

__global__ void silu_mul_cast(const float* __restrict__ a1, const float* __restrict__ a3,
                              bf16* __restrict__ out, long long n)
{
    long long i = (long long)blockIdx.x * blockDim.x + threadIdx.x;
    if (i < n) {
        float x = a1[i];
        float s = x / (1.f + __expf(-x));
        out[i] = f2bf(s * a3[i]);
    }
}

__global__ __launch_bounds__(256) void embed_gather(
    const int* __restrict__ ids, const float* __restrict__ embed,
    float* __restrict__ x, int D)
{
    int t = blockIdx.x;
    const float* src = embed + (long long)ids[t] * D;
    float* dst = x + (long long)t * D;
    for (int d = threadIdx.x; d < D; d += blockDim.x) dst[d] = src[d];
}

__global__ __launch_bounds__(256) void rmsnorm_k(
    const float* __restrict__ x, const float* __restrict__ wgt,
    float* __restrict__ yf, bf16* __restrict__ yb, int D)
{
    __shared__ float red[256];
    int t = blockIdx.x;
    const float* row = x + (long long)t * D;
    float ss = 0.f;
    for (int d = threadIdx.x; d < D; d += 256) { float v = row[d]; ss += v * v; }
    red[threadIdx.x] = ss; __syncthreads();
    for (int s = 128; s > 0; s >>= 1) {
        if (threadIdx.x < s) red[threadIdx.x] += red[threadIdx.x + s];
        __syncthreads();
    }
    float inv = rsqrtf(red[0] / D + 1e-6f);
    for (int d = threadIdx.x; d < D; d += 256) {
        float v = row[d] * inv * wgt[d];
        if (yf) yf[(long long)t * D + d] = v;
        if (yb) yb[(long long)t * D + d] = f2bf(v);
    }
}

// y[t, 0..15] = sigmoid(x[t,:] @ W[D,16] + bias)
__global__ __launch_bounds__(256) void proj16_sigmoid(
    const float* __restrict__ X, const float* __restrict__ W,
    const float* __restrict__ bias, float* __restrict__ Y, int D)
{
    __shared__ float red[16][17];
    int t = blockIdx.x;
    int j = threadIdx.x & 15, g = threadIdx.x >> 4;
    float p = 0.f;
    for (int k = g; k < D; k += 16) p += X[(long long)t * D + k] * W[k * 16 + j];
    red[g][j] = p; __syncthreads();
    if (threadIdx.x < 16) {
        float s = 0.f;
        for (int g2 = 0; g2 < 16; ++g2) s += red[g2][threadIdx.x];
        s += bias[threadIdx.x];
        Y[t * 16 + threadIdx.x] = 1.f / (1.f + __expf(-s));
    }
}

__global__ __launch_bounds__(64) void knorm_k(const float* __restrict__ kin, float* __restrict__ kout)
{
    __shared__ float red[64];
    int t = blockIdx.x, h = blockIdx.y;
    long long base = (long long)t * 1024 + h * 64;
    float v = kin[base + threadIdx.x];
    red[threadIdx.x] = v * v; __syncthreads();
    for (int s = 32; s > 0; s >>= 1) {
        if (threadIdx.x < s) red[threadIdx.x] += red[threadIdx.x + s];
        __syncthreads();
    }
    float inv = 1.f / fmaxf(sqrtf(red[0]), 1e-12f);
    kout[base + threadIdx.x] = v * inv;
}

// vt[(b,h,d), t] = vv[(b*T+t), h*64+d]   (bf16 -> bf16)
__global__ __launch_bounds__(256) void v_transpose(
    const bf16* __restrict__ vv, bf16* __restrict__ vt, int T)
{
    int t = blockIdx.x * 256 + threadIdx.x;
    int d = blockIdx.y, bh = blockIdx.z, b = bh >> 4, h = bh & 15;
    vt[((long long)bh * 64 + d) * T + t] =
        vv[((long long)b * T + t) * 1024 + h * 64 + d];
}

__global__ void gate_mean(const float* __restrict__ gate, float* __restrict__ gm, int n)
{
    int t = blockIdx.x * blockDim.x + threadIdx.x;
    if (t < n) {
        float s = 0.f;
        for (int j = 0; j < 16; ++j) s += gate[t * 16 + j];
        gm[t] = s * (1.f / 16.f);
    }
}

__global__ void swa_combine(float* __restrict__ x, const float* __restrict__ lo,
                            const float* __restrict__ ret, const float* __restrict__ gm,
                            int D, long long n)
{
    long long i = (long long)blockIdx.x * blockDim.x + threadIdx.x;
    if (i < n) {
        int row = (int)(i / D);
        x[i] += lo[i] + gm[row] * ret[i];
    }
}

// ---------------------------------------------------------------------------
// Host orchestration
// ---------------------------------------------------------------------------
extern "C" void kernel_launch(void* const* d_in, const int* in_sizes, int n_in,
                              void* d_out, int out_size, void* d_ws, size_t ws_size,
                              hipStream_t stream)
{
    (void)in_sizes; (void)n_in; (void)out_size; (void)ws_size;
    const int B = 2, T = 1024, D = 1024, H = 16, NT = B * T, HID = 2752, VOC = 32000;
    const float SCALE = 0.125f;  // 64^-0.5

    const int*   ids      = (const int*)  d_in[0];
    const float* embed    = (const float*)d_in[1];
    const float* g_k_w    = (const float*)d_in[2];
    const float* g_v_w    = (const float*)d_in[3];
    const float* g_o_w    = (const float*)d_in[4];
    const float* g_beta_w = (const float*)d_in[5];
    const float* g_beta_b = (const float*)d_in[6];
    const float* g_g_w    = (const float*)d_in[7];
    const float* g_g_b    = (const float*)d_in[8];
    const float* g_norm_w = (const float*)d_in[9];
    const float* s_q_w    = (const float*)d_in[10];
    const float* s_k_w    = (const float*)d_in[11];
    const float* s_v_w    = (const float*)d_in[12];
    const float* s_o_w    = (const float*)d_in[13];
    const float* s_gq_w   = (const float*)d_in[14];
    const float* s_ro_w   = (const float*)d_in[15];
    const float* s_gate_w = (const float*)d_in[16];
    const float* s_gate_b = (const float*)d_in[17];
    const float* s_norm_w = (const float*)d_in[18];
    const float* f0_w1    = (const float*)d_in[19];
    const float* f0_w3    = (const float*)d_in[20];
    const float* f0_w2    = (const float*)d_in[21];
    const float* f0_norm  = (const float*)d_in[22];
    const float* f1_w1    = (const float*)d_in[23];
    const float* f1_w3    = (const float*)d_in[24];
    const float* f1_w2    = (const float*)d_in[25];
    const float* f1_norm  = (const float*)d_in[26];
    const float* normf_w  = (const float*)d_in[27];
    float* logits = (float*)d_out;

    // bump allocator over workspace
    char* wp = (char*)d_ws;
    auto alloc = [&](size_t bytes) -> void* {
        void* r = (void*)wp;
        wp += (bytes + 255) & ~(size_t)255;
        return r;
    };
    const size_t ND = (size_t)NT * D, NH = (size_t)NT * HID;

    bf16* ebf   = (bf16*)alloc((size_t)VOC * D * 2);
    bf16* gkT   = (bf16*)alloc((size_t)D * D * 2);
    bf16* gvT   = (bf16*)alloc((size_t)D * D * 2);
    bf16* goT   = (bf16*)alloc((size_t)D * D * 2);
    bf16* sqT   = (bf16*)alloc((size_t)D * D * 2);
    bf16* skT   = (bf16*)alloc((size_t)D * D * 2);
    bf16* svT   = (bf16*)alloc((size_t)D * D * 2);
    bf16* soT   = (bf16*)alloc((size_t)D * D * 2);
    bf16* sgqT  = (bf16*)alloc((size_t)D * D * 2);
    bf16* sroT  = (bf16*)alloc((size_t)D * D * 2);
    bf16* f0w1T = (bf16*)alloc((size_t)HID * D * 2);
    bf16* f0w3T = (bf16*)alloc((size_t)HID * D * 2);
    bf16* f0w2T = (bf16*)alloc((size_t)D * HID * 2);
    bf16* f1w1T = (bf16*)alloc((size_t)HID * D * 2);
    bf16* f1w3T = (bf16*)alloc((size_t)HID * D * 2);
    bf16* f1w2T = (bf16*)alloc((size_t)D * HID * 2);

    float* xf     = (float*)alloc(ND * 4);
    float* xnf    = (float*)alloc(ND * 4);
    bf16*  xnbf   = (bf16*) alloc(ND * 2);
    float* krawf  = (float*)alloc(ND * 4);
    float* vrawf  = (float*)alloc(ND * 4);
    float* knf    = (float*)alloc(ND * 4);
    float* betaf  = (float*)alloc((size_t)NT * 16 * 4);
    float* gvecf  = (float*)alloc((size_t)NT * 16 * 4);
    bf16*  outsbf = (bf16*) alloc(ND * 2);
    bf16*  stbf   = (bf16*) alloc((size_t)B * H * 64 * 64 * 2);
    float* a1f    = (float*)alloc(NH * 4);
    float* a3f    = (float*)alloc(NH * 4);
    bf16*  hhbf   = (bf16*) alloc(NH * 2);
    bf16*  qbf2   = (bf16*) alloc(ND * 2);
    bf16*  kbf2   = (bf16*) alloc(ND * 2);
    bf16*  vvbf   = (bf16*) alloc(ND * 2);
    bf16*  vtbf   = (bf16*) alloc(((size_t)B * H * 64 * T + 128) * 2);
    bf16*  qgbf   = (bf16*) alloc(ND * 2);
    float* gatef  = (float*)alloc((size_t)NT * 16 * 4);
    float* gmf    = (float*)alloc((size_t)NT * 4);
    bf16*  locbf  = (bf16*) alloc(ND * 2);
    float* locof  = (float*)alloc(ND * 4);
    bf16*  retbf  = (bf16*) alloc(ND * 2);
    float* retf   = (float*)alloc(ND * 4);
    bf16*  finbf  = (bf16*) alloc(ND * 2);

    auto gemm = [&](const bf16* A, const bf16* Bt, float* C, bf16* Cbf,
                    const float* add, int M, int N, int K,
                    int lda, int ldb, int ldc, int relu = 0) {
        dim3 g(N / 64, M / 256, 1);
        gemm_bf16_wmma<<<g, 256, 0, stream>>>(A, Bt, C, Cbf, add, M, N, K,
                                              lda, ldb, ldc, 0, 0, 0, 0, 0, 0, 0, relu);
    };
    auto tcast = [&](const float* W, bf16* Wt, int K, int N) {
        transpose_cast<<<dim3((K + 31) / 32, (N + 31) / 32), dim3(32, 8), 0, stream>>>(W, Wt, K, N);
    };

    // ---- prep: bf16 weights ----
    cast_f32_bf16<<<(unsigned)(((long long)VOC * D + 255) / 256), 256, 0, stream>>>(
        embed, ebf, (long long)VOC * D);
    tcast(g_k_w, gkT, D, D);   tcast(g_v_w, gvT, D, D);   tcast(g_o_w, goT, D, D);
    tcast(s_q_w, sqT, D, D);   tcast(s_k_w, skT, D, D);   tcast(s_v_w, svT, D, D);
    tcast(s_o_w, soT, D, D);   tcast(s_gq_w, sgqT, D, D); tcast(s_ro_w, sroT, D, D);
    tcast(f0_w1, f0w1T, D, HID); tcast(f0_w3, f0w3T, D, HID); tcast(f0_w2, f0w2T, HID, D);
    tcast(f1_w1, f1w1T, D, HID); tcast(f1_w3, f1w3T, D, HID); tcast(f1_w2, f1w2T, HID, D);

    // ---- embedding ----
    embed_gather<<<NT, 256, 0, stream>>>(ids, embed, xf, D);

    // ---- GDN layer ----
    rmsnorm_k<<<NT, 256, 0, stream>>>(xf, g_norm_w, xnf, xnbf, D);
    gemm(xnbf, gkT, krawf, nullptr, nullptr, NT, D, D, D, D, D);
    gemm(xnbf, gvT, vrawf, nullptr, nullptr, NT, D, D, D, D, D);
    proj16_sigmoid<<<NT, 256, 0, stream>>>(xnf, g_beta_w, g_beta_b, betaf, D);
    proj16_sigmoid<<<NT, 256, 0, stream>>>(xnf, g_g_w, g_g_b, gvecf, D);
    knorm_k<<<dim3(NT, H), 64, 0, stream>>>(krawf, knf);
    gdn_scan<<<B * H, 64, 0, stream>>>(knf, vrawf, betaf, gvecf, outsbf, stbf, T);
    gemm(outsbf, goT, xf, nullptr, xf, NT, D, D, D, D, D);   // x += outs @ g_o_w

    // ---- FFN0 ----
    rmsnorm_k<<<NT, 256, 0, stream>>>(xf, f0_norm, nullptr, xnbf, D);
    gemm(xnbf, f0w1T, a1f, nullptr, nullptr, NT, HID, D, D, D, HID);
    gemm(xnbf, f0w3T, a3f, nullptr, nullptr, NT, HID, D, D, D, HID);
    silu_mul_cast<<<(unsigned)((NH + 255) / 256), 256, 0, stream>>>(a1f, a3f, hhbf, (long long)NH);
    gemm(hhbf, f0w2T, xf, nullptr, xf, NT, D, HID, HID, HID, D);

    // ---- SWA layer ----
    rmsnorm_k<<<NT, 256, 0, stream>>>(xf, s_norm_w, xnf, xnbf, D);
    gemm(xnbf, sqT, nullptr, qbf2, nullptr, NT, D, D, D, D, D);
    gemm(xnbf, skT, nullptr, kbf2, nullptr, NT, D, D, D, D, D);
    gemm(xnbf, svT, nullptr, vvbf, nullptr, NT, D, D, D, D, D);
    v_transpose<<<dim3(T / 256, 64, B * H), 256, 0, stream>>>(vvbf, vtbf, T);
    gemm(xnbf, sgqT, nullptr, qgbf, nullptr, NT, D, D, D, D, D, /*relu=*/1);
    proj16_sigmoid<<<NT, 256, 0, stream>>>(xnf, s_gate_w, s_gate_b, gatef, D);
    gate_mean<<<(NT + 255) / 256, 256, 0, stream>>>(gatef, gmf, NT);

    swa_attn<<<dim3(T / 64, H, B), 128, 0, stream>>>(qbf2, kbf2, vtbf, locbf, T, H, SCALE);
    gemm(locbf, soT, locof, nullptr, nullptr, NT, D, D, D, D, D);

    // retrieval: per (b,h)  retr[t, h*64+v] = qg[t, h*64+:] @ S^T[v,:]
    {
        dim3 g(1, T / 256, B * H);
        gemm_bf16_wmma<<<g, 256, 0, stream>>>(
            qgbf, stbf, nullptr, retbf, nullptr,
            T, 64, 64, D, 64, D,
            (long long)T * D, 64,              // A: batch stride, head stride
            (long long)H * 64 * 64, 64 * 64,   // Bt (S^T per b,h)
            (long long)T * D, 64,              // C
            H, 0);
    }
    gemm(retbf, sroT, retf, nullptr, nullptr, NT, D, D, D, D, D);
    swa_combine<<<(unsigned)((ND + 255) / 256), 256, 0, stream>>>(xf, locof, retf, gmf, D, (long long)ND);

    // ---- FFN1 ----
    rmsnorm_k<<<NT, 256, 0, stream>>>(xf, f1_norm, nullptr, xnbf, D);
    gemm(xnbf, f1w1T, a1f, nullptr, nullptr, NT, HID, D, D, D, HID);
    gemm(xnbf, f1w3T, a3f, nullptr, nullptr, NT, HID, D, D, D, HID);
    silu_mul_cast<<<(unsigned)((NH + 255) / 256), 256, 0, stream>>>(a1f, a3f, hhbf, (long long)NH);
    gemm(hhbf, f1w2T, xf, nullptr, xf, NT, D, HID, HID, HID, D);

    // ---- final norm + tied LM head (embed already is Bt = [VOCAB, D]) ----
    rmsnorm_k<<<NT, 256, 0, stream>>>(xf, normf_w, nullptr, finbf, D);
    gemm(finbf, ebf, logits, nullptr, nullptr, NT, VOC, D, D, D, VOC);
}
